// PAMNet_77257871720634
// MI455X (gfx1250) — matrix operand order repaired and checked by reading, not compiled, not run on hardware
//
#include <hip/hip_runtime.h>
#include <math.h>

// ---------------- model constants ----------------
#define DIMC 128
#define TDIM 16
#define ODIM 12
#define ADIM 9
#define DD   153          // feature dim
#define DP   160          // padded feature stride (10 x 16)
#define NRBF 16
#define NSPH 7
#define NRAD 6
#define CUTF 16.0f

typedef __bf16 bf16_t;
typedef bf16_t v16bf __attribute__((ext_vector_type(16)));
typedef float  v8f   __attribute__((ext_vector_type(8)));

// spherical Bessel zeros j_l, first 6, l = 0..6 (precomputed offline)
__constant__ float ZTAB[NSPH][NRAD] = {
  { 3.14159265f,  6.28318531f,  9.42477796f, 12.56637061f, 15.70796327f, 18.84955592f},
  { 4.49340946f,  7.72525184f, 10.90412166f, 14.06619391f, 17.22075527f, 20.37130296f},
  { 5.76345920f,  9.09501133f, 12.32294097f, 15.51460301f, 18.68903635f, 21.85387422f},
  { 6.98793200f, 10.41711855f, 13.69802315f, 16.92362129f, 20.12180617f, 23.30424717f},
  { 8.18256145f, 11.70490715f, 15.03966471f, 18.30125596f, 21.52541790f, 24.72756573f},
  { 9.35581211f, 12.96653017f, 16.35470964f, 19.65315286f, 22.90455060f, 26.12773515f},
  {10.51283541f, 14.20739209f, 17.64797495f, 20.98346482f, 24.26276804f, 27.50786836f}
};

// ---------------- small device helpers ----------------
__device__ __forceinline__ unsigned short f2bf(float f) {
  unsigned u = __float_as_uint(f);
  u += 0x7fffu + ((u >> 16) & 1u);           // RNE to bf16
  return (unsigned short)(u >> 16);
}
__device__ __forceinline__ unsigned pack2(float a, float b) {
  return (unsigned)f2bf(a) | ((unsigned)f2bf(b) << 16);
}

enum { ACT_NONE = 0, ACT_SILU = 1, ACT_GELU = 2 };

template<int ACT> __device__ __forceinline__ float actf(float v) {
  if (ACT == ACT_SILU) return v / (1.f + __expf(-v));
  if (ACT == ACT_GELU) {
    float c = v + 0.044715f * v * v * v;
    return 0.5f * v * (1.f + tanhf(0.7978845608f * c));
  }
  return v;
}

__device__ __forceinline__ float envel(float x) {   // DimeNet envelope p=6
  float x2 = x * x;
  float x5 = x2 * x2 * x;
  return 1.f / x - 28.f * x5 + 48.f * x5 * x - 21.f * x5 * x2;
}

__device__ __forceinline__ float sphj(int l, float x) {   // spherical Bessel j_l
  float xs = (fabsf(x) < 1e-8f) ? 1e-8f : x;
  float j0 = __sinf(xs) / xs;
  if (l == 0) return j0;
  float j1 = __sinf(xs) / (xs * xs) - __cosf(xs) / xs;
  float jm = j0, jc = j1;
  for (int ll = 1; ll < l; ++ll) {
    float jn = (2.f * ll + 1.f) / xs * jc - jm;
    jm = jc; jc = jn;
  }
  return jc;
}

// ---------------- weight pre-pack ----------------
// fp32 W[K,Nout] row-major  ->  packed bf16 pairs, column-major by n:
//   out[n * (Kp/2) + k/2] = {W[k,n], W[k+1,n]},  zero-padded to Kp and NTile.
// Done once per launch; makes GEMM B-staging an unconditional u32 copy and
// guarantees B==0 for k>=K / n>=Nout (so A staging needs no bounds logic).
__global__ void k_packB(const float* __restrict__ W, unsigned* __restrict__ out,
                        int K, int Nout, int NTile, int Kp) {
  int i = blockIdx.x * 256 + threadIdx.x;
  int half = Kp >> 1;
  if (i >= NTile * half) return;
  int n = i / half, kp = i % half;
  int k0 = kp * 2;
  float f0 = (k0     < K && n < Nout) ? W[(size_t)k0       * Nout + n] : 0.f;
  float f1 = (k0 + 1 < K && n < Nout) ? W[(size_t)(k0 + 1) * Nout + n] : 0.f;
  out[i] = pack2(f0, f1);
}

// ---------------- WMMA GEMM (the workhorse) ----------------
// C[M,Nout] = epilogue( act( A[M,K] @ B + bias ) ), B pre-packed bf16 (zero-padded).
// A is a virtual concat of up to NSEG gathered segments (compile-time):
//   k <  W0        -> A0[gidx0(r)][k]            (GI0 -> gather via I0, else identity)
//   k <  W0+W1     -> A1[gidx1(r)][k-W0]         (GI1 -> gather via I1)
//   else           -> A2[r][k-W0-W1]             (identity)
// Staging is fully unconditional: rows clamped to M-1 (harmless, never stored),
// k runs to Kp with garbage reads neutralized by zero-padded B. Sources must
// have >= 32 floats of mapped slack past their end (workspace guarantees this).
template<int NT, int ACT, bool SCATTER, int NSEG, bool GI0, bool GI1,
         bool BIAS, bool MUL, bool MIDX, bool RES>
__global__ __launch_bounds__(256)
void wmma_gemm(const float* __restrict__ A0, const int* __restrict__ I0, int L0, int W0,
               const float* __restrict__ A1, const int* __restrict__ I1, int L1, int W1,
               const float* __restrict__ A2, int L2,
               int Kp,
               const unsigned* __restrict__ Bp, int Nout,
               const float* __restrict__ bias,
               const float* __restrict__ mul, const int* __restrict__ mulIdx, int ldmul,
               const float* __restrict__ resid, int ldres,
               float* __restrict__ C, const int* __restrict__ outIdx, int ldc,
               int M)
{
  __shared__ unsigned As[128][17];        // 128 rows x 16 packed-bf16 K-pairs (+pad)
  __shared__ unsigned Bs[NT * 16][17];    // col-major by n: [n][kpair] (+pad)

  const int tid = threadIdx.x;
  const int lane = tid & 31, wave = tid >> 5;
  const int rowBlock = blockIdx.x * 128;
  const int c1 = W0 + W1;
  const int halfK = Kp >> 1;

  v8f acc[NT];
  const v8f vzero = {0.f, 0.f, 0.f, 0.f, 0.f, 0.f, 0.f, 0.f};
#pragma unroll
  for (int t = 0; t < NT; ++t) acc[t] = vzero;

  for (int kb = 0; kb < Kp; kb += 32) {
    // ---- stage A tile: 128 rows x 16 pairs; all loads unconditional ----
#pragma unroll
    for (int it = 0; it < 8; ++it) {
      int e = tid + it * 256;
      int r = e >> 4, kp = e & 15;
      int grow = rowBlock + r;
      grow = (grow < M) ? grow : (M - 1);          // clamp; extra rows never stored
      const float* p0;
      const float* p1 = nullptr;
      const float* p2 = nullptr;
      { int rr = GI0 ? I0[grow] : grow; p0 = A0 + (size_t)rr * L0; }
      if (NSEG == 3) {
        int rr1 = GI1 ? I1[grow] : grow;
        p1 = A1 + (size_t)rr1 * L1;
        p2 = A2 + (size_t)grow * L2;
      }
      int k = kb + kp * 2;
      float f0, f1;
      if (NSEG == 1) {
        f0 = p0[k];
        f1 = p0[k + 1];
        __builtin_prefetch(p0 + k + 32, 0, 1);     // next k-step (global_prefetch_b8)
      } else {
        const float* pa = (k < W0) ? (p0 + k)
                        : ((k < c1) ? (p1 + (k - W0)) : (p2 + (k - c1)));
        int k1 = k + 1;
        const float* pb = (k1 < W0) ? (p0 + k1)
                        : ((k1 < c1) ? (p1 + (k1 - W0)) : (p2 + (k1 - c1)));
        f0 = *pa;
        f1 = *pb;
        __builtin_prefetch(pa + 32, 0, 1);
      }
      As[r][kp] = pack2(f0, f1);
    }
    // ---- stage B tile: unconditional copy of pre-packed pairs ----
#pragma unroll
    for (int it = 0; it < NT; ++it) {
      int e = tid + it * 256;
      int n = e >> 4, kp = e & 15;
      Bs[n][kp] = Bp[(size_t)n * halfK + (kb >> 1) + kp];
    }
    __syncthreads();

    // ---- fragments + WMMA ----
    // A 16x32 bf16 layout: lanes 0-15 M=lane, K-pairs {0..3, 8..11}; lanes 16-31 +4
    union Frag { unsigned u[8]; v16bf v; };
    Frag af;
    {
      int mrow = wave * 16 + (lane & 15);
      int hi = (lane >= 16) ? 4 : 0;
#pragma unroll
      for (int p = 0; p < 8; ++p) {
        int kp = (p < 4 ? p : p + 4) + hi;
        af.u[p] = As[mrow][kp];
      }
    }
#pragma unroll
    for (int t = 0; t < NT; ++t) {
      // B 32x16 bf16 layout: lanes 0-15 N=lane, K-pairs {0..3, 8..11}; lanes 16-31 +4
      Frag bf;
      int ncol = t * 16 + (lane & 15);
      int hk = (lane >= 16) ? 4 : 0;
#pragma unroll
      for (int p = 0; p < 8; ++p) bf.u[p] = Bs[ncol][(p < 4 ? p : p + 4) + hk];
      acc[t] = __builtin_amdgcn_wmma_f32_16x16x32_bf16(
          false, af.v, false, bf.v, (short)0, acc[t], false, false);
    }
    __syncthreads();
  }

  // ---- epilogue ----
  // C/D 16x16 f32 layout: VGPR r -> M=r (lanes 0-15) / M=r+8 (lanes 16-31), N=lane&15
  int rbase = rowBlock + wave * 16 + ((lane < 16) ? 0 : 8);
#pragma unroll
  for (int t = 0; t < NT; ++t) {
    int col = t * 16 + (lane & 15);
    if (col >= Nout) continue;
#pragma unroll
    for (int r = 0; r < 8; ++r) {
      int row = rbase + r;
      if (row >= M) continue;
      float v = acc[t][r];
      if (BIAS) v += bias[col];
      v = actf<ACT>(v);
      if (MUL) {
        int mr = MIDX ? mulIdx[row] : row;
        v *= mul[(size_t)mr * ldmul + col];
      }
      if (SCATTER) {
        atomicAdd(&C[(size_t)outIdx[row] * ldc + col], v);
      } else {
        if (RES) v += resid[(size_t)row * ldres + col];
        C[(size_t)row * ldc + col] = v;
      }
    }
  }
}

// ---------------- elementwise kernels ----------------
__global__ void k_copy(const float* __restrict__ a, float* __restrict__ b, int n) {
  int i = blockIdx.x * 256 + threadIdx.x;
  if (i < n) b[i] = a[i];
}

__global__ void k_timefeat(const float* __restrict__ t, float* __restrict__ te, int N) {
  int i = blockIdx.x * 256 + threadIdx.x;
  if (i >= N * DIMC) return;
  int n = i / DIMC, c = i % DIMC;
  float f = __expf((float)(c & 63) * (-9.21034037198f / 63.f));   // -log(1e4)/(half-1)
  float v = t[n] * f;
  te[i] = (c < 64) ? __sinf(v) : __cosf(v);
}

// x[:,128..136]=feat, x[:,137..152]=temb, x[:,153..159]=0
__global__ void k_xtail(const float* __restrict__ feat, const float* __restrict__ temb,
                        float* __restrict__ x, int N) {
  int i = blockIdx.x * 256 + threadIdx.x;
  if (i >= N * 32) return;
  int n = i / 32, c = i % 32;
  float v = 0.f;
  if (c < ADIM)             v = feat[n * ADIM + c];
  else if (c < ADIM + TDIM) v = temb[n * TDIM + (c - ADIM)];
  x[(size_t)n * DP + DIMC + c] = v;
}

__global__ void k_dist(const float* __restrict__ pos, const int* __restrict__ src,
                       const int* __restrict__ dst, float* __restrict__ dist, int E) {
  int e = blockIdx.x * 256 + threadIdx.x;
  if (e >= E) return;
  int a = src[e], b = dst[e];
  float dx = pos[b * 3 + 0] - pos[a * 3 + 0];
  float dy = pos[b * 3 + 1] - pos[a * 3 + 1];
  float dz = pos[b * 3 + 2] - pos[a * 3 + 2];
  dist[e] = sqrtf(dx * dx + dy * dy + dz * dz);
}

// rbf features [E,32]: 0..15 envelope*sin(freq*x), 16..18 edge_attr, rest 0
__global__ void k_rbf(const float* __restrict__ dist, const float* __restrict__ ea,
                      const float* __restrict__ freq, float* __restrict__ out, int E) {
  int i = blockIdx.x * 256 + threadIdx.x;
  if (i >= E * 32) return;
  int e = i / 32, c = i % 32;
  float v = 0.f;
  if (c < NRBF) {
    float x = dist[e] / CUTF;
    v = envel(x) * __sinf(freq[c] * x);
  } else if (c < NRBF + 3) {
    v = ea[e * 3 + (c - NRBF)];
  }
  out[i] = v;
}

// radial Bessel basis per local edge: [E,48], 42 used
__global__ void k_rad(const float* __restrict__ dist, float* __restrict__ rad, int E) {
  int i = blockIdx.x * 256 + threadIdx.x;
  if (i >= E * 42) return;
  int e = i / 42, c = i % 42, l = c / 6, n = c % 6;
  float x = dist[e] / CUTF;
  float z = ZTAB[l][n];
  float nrm = 1.41421356f / fabsf(sphj(l + 1, z));
  rad[(size_t)e * 48 + c] = envel(x) * nrm * sphj(l, z * x);
}

// angle = atan2(|v1 x v2|, v1.v2), v1 = pos[b]-pos[a], v2 = pos[c]-pos[b]
__global__ void k_angle(const float* __restrict__ pos, const int* __restrict__ ia,
                        const int* __restrict__ ib, const int* __restrict__ ic,
                        float* __restrict__ ang, int T) {
  int t = blockIdx.x * 256 + threadIdx.x;
  if (t >= T) return;
  int a = ia[t], b = ib[t], c = ic[t];
  float v1x = pos[b*3+0]-pos[a*3+0], v1y = pos[b*3+1]-pos[a*3+1], v1z = pos[b*3+2]-pos[a*3+2];
  float v2x = pos[c*3+0]-pos[b*3+0], v2y = pos[c*3+1]-pos[b*3+1], v2z = pos[c*3+2]-pos[b*3+2];
  float cx = v1y*v2z - v1z*v2y, cy = v1z*v2x - v1x*v2z, cz = v1x*v2y - v1y*v2x;
  ang[t] = atan2f(sqrtf(cx*cx + cy*cy + cz*cz), v1x*v2x + v1y*v2y + v1z*v2z);
}

// sbf features [T,48] = rad[idx_edge] * Y_l * P_l(cos angle)
__global__ void k_sbf(const float* __restrict__ rad, const float* __restrict__ ang,
                      const int* __restrict__ idxe, float* __restrict__ out, int T) {
  int i = blockIdx.x * 256 + threadIdx.x;
  if (i >= T * 48) return;
  int t = i / 48, c = i % 48;
  if (c >= 42) { out[i] = 0.f; return; }
  int l = c / 6;
  float cs = __cosf(ang[t]);
  float P = 1.f;
  if (l >= 1) {
    float pm = 1.f, pc = cs;
    for (int ll = 1; ll < l; ++ll) {
      float pn = ((2.f * ll + 1.f) * cs * pc - (float)ll * pm) / (float)(ll + 1);
      pm = pc; pc = pn;
    }
    P = pc;
  }
  float yl = sqrtf((2.f * l + 1.f) * 0.0795774715f);   // sqrt((2l+1)/4pi)
  out[i] = rad[(size_t)idxe[t] * 48 + c] * yl * P;
}

// attention softmax over 4 heads + final linear (40 -> 12)
__global__ void k_head(const float* og0, const float* ag0, const float* ol0, const float* al0,
                       const float* og1, const float* ag1, const float* ol1, const float* al1,
                       const float* __restrict__ temb, const float* __restrict__ W,
                       const float* __restrict__ b, float* __restrict__ out, int N) {
  int n = blockIdx.x * 256 + threadIdx.x;
  if (n >= N) return;
  float o[2][24], a[2][24];
  for (int c = 0; c < ODIM; ++c) {
    o[0][c] = og0[n*ODIM+c]; o[0][ODIM+c] = ol0[n*ODIM+c];
    o[1][c] = og1[n*ODIM+c]; o[1][ODIM+c] = ol1[n*ODIM+c];
    a[0][c] = ag0[n*ODIM+c]; a[0][ODIM+c] = al0[n*ODIM+c];
    a[1][c] = ag1[n*ODIM+c]; a[1][ODIM+c] = al1[n*ODIM+c];
  }
  float s24[24];
  for (int c = 0; c < 24; ++c) s24[c] = 0.f;
  for (int L = 0; L < 2; ++L) {
    float mx = -1e30f;
    for (int c = 0; c < 24; ++c) {
      float v = a[L][c];
      v = (v >= 0.f) ? v : 0.2f * v;               // leaky_relu(0.2)
      a[L][c] = v;
      mx = fmaxf(mx, v);
    }
    float s = 0.f;
    for (int c = 0; c < 24; ++c) { a[L][c] = __expf(a[L][c] - mx); s += a[L][c]; }
    float inv = 1.f / s;
    for (int c = 0; c < 24; ++c) s24[c] += o[L][c] * a[L][c] * inv;
  }
  float acc[ODIM];
  for (int j = 0; j < ODIM; ++j) acc[j] = b[j];
  for (int c = 0; c < 24; ++c)
    for (int j = 0; j < ODIM; ++j) acc[j] += s24[c] * W[c * ODIM + j];
  for (int k = 0; k < TDIM; ++k) {
    float tv = temb[n * TDIM + k];
    for (int j = 0; j < ODIM; ++j) acc[j] += tv * W[(24 + k) * ODIM + j];
  }
  for (int j = 0; j < ODIM; ++j) out[n * ODIM + j] = acc[j];
}

// ---------------- host-side launch helpers ----------------
static inline int cdiv(int a, int b) { return (a + b - 1) / b; }

template<int NT, int ACT, bool SC, int NSEG, bool GI0, bool GI1,
         bool BIAS, bool MUL, bool MIDX, bool RES>
static void G(hipStream_t st,
              const float* A0, const int* I0, int L0, int W0,
              const float* A1, const int* I1, int L1, int W1,
              const float* A2, int L2,
              int Kp, const unsigned* Bp, int Nout, const float* bias,
              const float* mul, const int* mulIdx, int ldmul,
              const float* resid, int ldres,
              float* C, const int* outIdx, int ldc, int M) {
  wmma_gemm<NT, ACT, SC, NSEG, GI0, GI1, BIAS, MUL, MIDX, RES>
      <<<dim3(cdiv(M, 128)), dim3(256), 0, st>>>(
      A0, I0, L0, W0, A1, I1, L1, W1, A2, L2, Kp, Bp, Nout, bias,
      mul, mulIdx, ldmul, resid, ldres, C, outIdx, ldc, M);
}

extern "C" void kernel_launch(void* const* d_in, const int* in_sizes, int n_in,
                              void* d_out, int out_size, void* d_ws, size_t ws_size,
                              hipStream_t stream) {
  (void)n_in; (void)ws_size; (void)out_size;
  auto F = [&](int i) { return (const float*)d_in[i]; };
  auto I = [&](int i) { return (const int*)d_in[i]; };

  // ---- inputs (setup_inputs order; params leaves in JAX sorted-key order) ----
  const float* pos  = F(0);
  const float* feat = F(1);
  const float* tt   = F(2);
  const float* eaG  = F(3);
  const float* eaL  = F(4);
  const int N  = in_sizes[0] / 3;
  const int EG = in_sizes[3] / 3;
  const int EL = in_sizes[4] / 3;

  const float* freq_g = F(5);
  const float* freq_l = F(6);
  const int GB[2] = {7, 28};    // global layer param bases (21 leaves each)
  const int IW = 49, IB = 50;   // init
  const int LB[2] = {51, 85};   // local layer param bases (34 leaves each)
  const int RBG = 119, RBL = 121, SB1 = 123, SB2 = 125, OL = 127, T1W = 129, T2W = 131;

  const int* eig = I(133); const int* j_g = eig;        const int* i_g = eig + EG;
  const int* eil = I(134); const int* j_l = eil;        const int* i_l = eil + EL;
  const int* idx_i  = I(135); const int* idx_j  = I(136); const int* idx_k = I(137);
  const int* idx_kj = I(138); const int* idx_ji = I(139);
  const int* idx_ip = I(140); const int* idx_j1 = I(141); const int* idx_j2 = I(142);
  const int* idx_jjp = I(143); const int* idx_jip = I(144);
  const int T2 = in_sizes[138];
  const int T1 = in_sizes[143];

  // ---- workspace bump allocator (256B aligned; every buffer has tail slack) ----
  size_t off = 0;
  auto alloc = [&](size_t nelem) -> float* {
    float* p = (float*)((char*)d_ws + off);
    off = (off + nelem * 4 + 255) & ~(size_t)255;
    return p;
  };
  float* posp  = alloc((size_t)N * 3 + 64);    // padded copy of pos (A-source slack)
  float* te128 = alloc((size_t)N * DIMC);
  float* teh   = alloc((size_t)N * TDIM + 64);
  float* temb  = alloc((size_t)N * TDIM);
  float* X = alloc((size_t)N * DP);
  float* H = alloc((size_t)N * DP);
  float* T = alloc((size_t)N * DP);
  float* Gb = alloc((size_t)N * DP);
  float* distg = alloc(EG);
  float* distl = alloc(EL);
  float* rbffg = alloc((size_t)EG * 32);
  float* rbffl = alloc((size_t)EL * 32);
  float* eaRg  = alloc((size_t)EG * DP);
  float* eaRl  = alloc((size_t)EL * DP);
  float* radl  = alloc((size_t)EL * 48);
  float* ang2  = alloc(T2);
  float* ang1  = alloc(T1);
  float* sbff2 = alloc((size_t)T2 * 48 + 64);
  float* sbff1 = alloc((size_t)T1 * 48 + 64);
  float* eaS2  = alloc((size_t)T2 * DP);
  float* eaS1  = alloc((size_t)T1 * DP);
  float* MKJ   = alloc((size_t)EL * DP);
  float* MJJ   = alloc((size_t)EL * DP);
  float* MACC  = alloc((size_t)EL * DP);
  float* R     = alloc((size_t)EL * DP);
  float* OUTS  = alloc((size_t)8 * N * ODIM);
  float* og[2] = {OUTS + 0 * N * ODIM, OUTS + 4 * N * ODIM};
  float* ag[2] = {OUTS + 1 * N * ODIM, OUTS + 5 * N * ODIM};
  float* ol[2] = {OUTS + 2 * N * ODIM, OUTS + 6 * N * ODIM};
  float* al[2] = {OUTS + 3 * N * ODIM, OUTS + 7 * N * ODIM};

  // pack a weight leaf [K,Nout] -> padded bf16 pairs for an NT-tile GEMM
  auto packW = [&](int leaf, int K, int Nout, int NT) -> const unsigned* {
    int Kp = cdiv(K, 32) * 32;
    int half = Kp / 2, NTile = NT * 16;
    unsigned* p = (unsigned*)alloc((size_t)NTile * half);
    k_packB<<<cdiv(NTile * half, 256), 256, 0, stream>>>(F(leaf), p, K, Nout, NTile, Kp);
    return p;
  };

  const int* NI = nullptr;
  const float* NF = nullptr;

  k_copy<<<cdiv(N * 3, 256), 256, 0, stream>>>(pos, posp, N * 3);

  // ---- time embedding ----
  k_timefeat<<<cdiv(N * DIMC, 256), 256, 0, stream>>>(tt, te128, N);
  G<1, ACT_GELU, false, 1, false, false, true, false, false, false>(
      stream, te128, NI, DIMC, DIMC, NF, NI, 0, 0, NF, 0,
      128, packW(T1W, DIMC, TDIM, 1), TDIM, F(T1W + 1), NF, NI, 0, NF, 0,
      teh, NI, TDIM, N);
  G<1, ACT_NONE, false, 1, false, false, true, false, false, false>(
      stream, teh, NI, TDIM, TDIM, NF, NI, 0, 0, NF, 0,
      32, packW(T2W, TDIM, TDIM, 1), TDIM, F(T2W + 1), NF, NI, 0, NF, 0,
      temb, NI, TDIM, N);

  // ---- x0 = concat(silu(pos@init), feat, temb) ----
  G<8, ACT_SILU, false, 1, false, false, true, false, false, false>(
      stream, posp, NI, 3, 3, NF, NI, 0, 0, NF, 0,
      32, packW(IW, 3, DIMC, 8), DIMC, F(IB), NF, NI, 0, NF, 0, X, NI, DP, N);
  k_xtail<<<cdiv(N * 32, 256), 256, 0, stream>>>(feat, temb, X, N);

  // ---- geometry features ----
  k_dist<<<cdiv(EG, 256), 256, 0, stream>>>(pos, j_g, i_g, distg, EG);
  k_dist<<<cdiv(EL, 256), 256, 0, stream>>>(pos, j_l, i_l, distl, EL);
  k_rbf<<<cdiv(EG * 32, 256), 256, 0, stream>>>(distg, eaG, freq_g, rbffg, EG);
  k_rbf<<<cdiv(EL * 32, 256), 256, 0, stream>>>(distl, eaL, freq_l, rbffl, EL);
  G<10, ACT_SILU, false, 1, false, false, true, false, false, false>(
      stream, rbffg, NI, 32, NRBF + 3, NF, NI, 0, 0, NF, 0,
      32, packW(RBG, NRBF + 3, DD, 10), DD, F(RBG + 1), NF, NI, 0, NF, 0,
      eaRg, NI, DP, EG);
  G<10, ACT_SILU, false, 1, false, false, true, false, false, false>(
      stream, rbffl, NI, 32, NRBF + 3, NF, NI, 0, 0, NF, 0,
      32, packW(RBL, NRBF + 3, DD, 10), DD, F(RBL + 1), NF, NI, 0, NF, 0,
      eaRl, NI, DP, EL);
  k_rad<<<cdiv(EL * 42, 256), 256, 0, stream>>>(distl, radl, EL);
  k_angle<<<cdiv(T2, 256), 256, 0, stream>>>(pos, idx_i, idx_j, idx_k, ang2, T2);
  k_angle<<<cdiv(T1, 256), 256, 0, stream>>>(pos, idx_ip, idx_j1, idx_j2, ang1, T1);
  k_sbf<<<cdiv(T2 * 48, 256), 256, 0, stream>>>(radl, ang2, idx_kj, sbff2, T2);
  k_sbf<<<cdiv(T1 * 48, 256), 256, 0, stream>>>(radl, ang1, idx_jjp, sbff1, T1);
  G<10, ACT_SILU, false, 1, false, false, true, false, false, false>(
      stream, sbff2, NI, 48, 42, NF, NI, 0, 0, NF, 0,
      64, packW(SB2, 42, DD, 10), DD, F(SB2 + 1), NF, NI, 0, NF, 0, eaS2, NI, DP, T2);
  G<10, ACT_SILU, false, 1, false, false, true, false, false, false>(
      stream, sbff1, NI, 48, 42, NF, NI, 0, 0, NF, 0,
      64, packW(SB1, 42, DD, 10), DD, F(SB1 + 1), NF, NI, 0, NF, 0, eaS1, NI, DP, T1);

  // dense D->D step, optionally with post-activation residual
  auto dense = [&](const float* in, int Wi, int bi, const float* res, float* out) {
    const unsigned* Bp = packW(Wi, DD, DD, 10);
    if (res)
      G<10, ACT_SILU, false, 1, false, false, true, false, false, true>(
          stream, in, NI, DP, DD, NF, NI, 0, 0, NF, 0,
          160, Bp, DD, F(bi), NF, NI, 0, res, DP, out, NI, DP, N);
    else
      G<10, ACT_SILU, false, 1, false, false, true, false, false, false>(
          stream, in, NI, DP, DD, NF, NI, 0, 0, NF, 0,
          160, Bp, DD, F(bi), NF, NI, 0, NF, 0, out, NI, DP, N);
  };

  // shared tail: res1, x2, res2, res3 (Gb -> X, using T/H temps)
  auto tail = [&](int r1, int x2Wi, int x2bi, int r2, int r3) {
    dense(Gb, r1, r1 + 1, nullptr, T);
    dense(T, r1 + 2, r1 + 3, Gb, X);
    dense(X, x2Wi, x2bi, X, T);
    float* tmp = X; X = T; T = tmp;
    for (int rb = 0; rb < 2; ++rb) {
      int base = (rb == 0) ? r2 : r3;
      dense(X, base, base + 1, nullptr, H);
      dense(H, base + 2, base + 3, X, T);
      tmp = X; X = T; T = tmp;
    }
  };

  for (int L = 0; L < 2; ++L) {
    // ================= GLOBAL layer =================
    int g = GB[L];
    dense(X, g + 17, g + 18, nullptr, H);                       // h = silu(x@x1)
    k_copy<<<cdiv(N * DP, 256), 256, 0, stream>>>(X, Gb, N * DP);
    // agg[i] += silu(concat(h[i],h[j],eaRg) @ m)   (fused gather + scatter)
    G<10, ACT_SILU, true, 3, true, true, true, false, false, false>(
        stream, H, i_g, DP, DD, H, j_g, DP, DD, eaRg, DP,
        480, packW(g + 3, 3 * DD, DD, 10), DD, F(g + 4), NF, NI, 0, NF, 0,
        Gb, i_g, DP, EG);
    tail(g + 5, g + 19, g + 20, g + 9, g + 13);
    G<1, ACT_NONE, false, 1, false, false, false, false, false, false>(
        stream, X, NI, DP, DD, NF, NI, 0, 0, NF, 0,
        160, packW(g + 2, DD, ODIM, 1), ODIM, NF, NF, NI, 0, NF, 0, og[L], NI, ODIM, N);
    G<1, ACT_NONE, false, 1, false, false, true, false, false, false>(
        stream, X, NI, DP, DD, NF, NI, 0, 0, NF, 0,
        160, packW(g + 0, DD, ODIM, 1), ODIM, F(g + 1), NF, NI, 0, NF, 0, ag[L], NI, ODIM, N);

    // ================= LOCAL layer =================
    int l = LB[L];
    dense(X, l + 30, l + 31, nullptr, H);                       // h = silu(x@x1)
    // m_kj = silu(concat @ kj) * (rbf @ lin_rbf1)
    G<10, ACT_NONE, false, 1, false, false, false, false, false, false>(
        stream, eaRl, NI, DP, DD, NF, NI, 0, 0, NF, 0,
        160, packW(l + 9, DD, DD, 10), DD, NF, NF, NI, 0, NF, 0, R, NI, DP, EL);
    G<10, ACT_SILU, false, 3, true, true, true, true, false, false>(
        stream, H, i_l, DP, DD, H, j_l, DP, DD, eaRl, DP,
        480, packW(l + 6, 3 * DD, DD, 10), DD, F(l + 7), R, NI, DP, NF, 0,
        MKJ, NI, DP, EL);
    // m_jj = silu(concat @ jj) * (rbf @ lin_rbf2)
    G<10, ACT_NONE, false, 1, false, false, false, false, false, false>(
        stream, eaRl, NI, DP, DD, NF, NI, 0, 0, NF, 0,
        160, packW(l + 10, DD, DD, 10), DD, NF, NF, NI, 0, NF, 0, R, NI, DP, EL);
    G<10, ACT_SILU, false, 3, true, true, true, true, false, false>(
        stream, H, i_l, DP, DD, H, j_l, DP, DD, eaRl, DP,
        480, packW(l + 4, 3 * DD, DD, 10), DD, F(l + 5), R, NI, DP, NF, 0,
        MJJ, NI, DP, EL);
    // macc = silu(concat @ ji)
    G<10, ACT_SILU, false, 3, true, true, true, false, false, false>(
        stream, H, i_l, DP, DD, H, j_l, DP, DD, eaRl, DP,
        480, packW(l + 2, 3 * DD, DD, 10), DD, F(l + 3), NF, NI, 0, NF, 0,
        MACC, NI, DP, EL);
    // macc[idx_ji] += silu(eaS2 @ sbf2) * m_kj[idx_kj]   (triplet GEMM, fused scatter)
    G<10, ACT_SILU, true, 1, false, false, true, true, true, false>(
        stream, eaS2, NI, DP, DD, NF, NI, 0, 0, NF, 0,
        160, packW(l + 28, DD, DD, 10), DD, F(l + 29), MKJ, idx_kj, DP, NF, 0,
        MACC, idx_ji, DP, T2);
    // macc[idx_ji_pair] += silu(eaS1 @ sbf1) * m_jj[idx_jj_pair]
    G<10, ACT_SILU, true, 1, false, false, true, true, true, false>(
        stream, eaS1, NI, DP, DD, NF, NI, 0, 0, NF, 0,
        160, packW(l + 26, DD, DD, 10), DD, F(l + 27), MJJ, idx_jjp, DP, NF, 0,
        MACC, idx_jip, DP, T1);
    // agg = x; agg[i] += silu(macc @ mm) * (rbf @ lin_rbf_out)
    G<10, ACT_NONE, false, 1, false, false, false, false, false, false>(
        stream, eaRl, NI, DP, DD, NF, NI, 0, 0, NF, 0,
        160, packW(l + 11, DD, DD, 10), DD, NF, NF, NI, 0, NF, 0, R, NI, DP, EL);
    k_copy<<<cdiv(N * DP, 256), 256, 0, stream>>>(X, Gb, N * DP);
    G<10, ACT_SILU, true, 1, false, false, true, true, false, false>(
        stream, MACC, NI, DP, DD, NF, NI, 0, 0, NF, 0,
        160, packW(l + 12, DD, DD, 10), DD, F(l + 13), R, NI, DP, NF, 0,
        Gb, i_l, DP, EL);
    tail(l + 14, l + 32, l + 33, l + 18, l + 22);
    G<1, ACT_NONE, false, 1, false, false, false, false, false, false>(
        stream, X, NI, DP, DD, NF, NI, 0, 0, NF, 0,
        160, packW(l + 8, DD, ODIM, 1), ODIM, NF, NF, NI, 0, NF, 0, ol[L], NI, ODIM, N);
    G<1, ACT_NONE, false, 1, false, false, true, false, false, false>(
        stream, X, NI, DP, DD, NF, NI, 0, 0, NF, 0,
        160, packW(l + 0, DD, ODIM, 1), ODIM, F(l + 1), NF, NI, 0, NF, 0, al[L], NI, ODIM, N);
  }

  // ---- attention head + final linear ----
  k_head<<<cdiv(N, 256), 256, 0, stream>>>(og[0], ag[0], ol[0], al[0],
                                           og[1], ag[1], ol[1], al[1],
                                           temb, F(OL), F(OL + 1), (float*)d_out, N);
}
